// TransformerBlock_19834158973048
// MI455X (gfx1250) — matrix operand order repaired and checked by reading, not compiled, not run on hardware
//
#include <hip/hip_runtime.h>
#include <stdint.h>

#define DIM 4096
#define N_HEADS 32
#define N_KV_HEADS 8
#define HEAD_DIM 128
#define SEQ 2048
#define HIDDEN 11008
#define KV_DIM (N_KV_HEADS * HEAD_DIM)   // 1024

typedef __bf16 v16bf __attribute__((ext_vector_type(16)));
typedef float  v8f   __attribute__((ext_vector_type(8)));

union Frag16 { v16bf v; uint32_t u[8]; };

__device__ __forceinline__ uint16_t f2bf(float f) {
  uint32_t u = __float_as_uint(f);
  uint32_t r = (u + 0x7FFFu + ((u >> 16) & 1u)) >> 16;
  return (uint16_t)r;
}
__device__ __forceinline__ uint32_t pack2bf(float a, float b) {
  return (uint32_t)f2bf(a) | ((uint32_t)f2bf(b) << 16);
}
// A-operand (16x32 bf16) K index for vgpr v, lane-half h
__device__ __forceinline__ int a_k(int v, int h) {
  return 16 * h + 2 * v;
}

// Async DMA: 16B per lane, global -> LDS. Tracked by ASYNCcnt.
__device__ __forceinline__ void async_b128(uint32_t lds_off, const void* gaddr) {
  asm volatile("global_load_async_to_lds_b128 %0, %1, off"
               :: "v"(lds_off), "v"(gaddr) : "memory");
}
#define WAIT_ASYNC(n) asm volatile("s_wait_asynccnt " #n ::: "memory")
__device__ __forceinline__ uint32_t lds_off32(const void* p) {
  return (uint32_t)(uintptr_t)p;  // shared aperture: low 32 bits = LDS offset
}

// ---------------------------------------------------------------------------
// Convert + transpose: out[c][r] = bf16(in[r][c]); out row stride = rows.
// ---------------------------------------------------------------------------
__global__ void __launch_bounds__(256)
transpose_bf16_kernel(const float* __restrict__ in, uint16_t* __restrict__ out,
                      int rows, int cols) {
  __shared__ float tile[32][33];
  const int bc = blockIdx.x * 32, br = blockIdx.y * 32;
  const int tx = threadIdx.x & 31, ty = threadIdx.x >> 5;   // 32 x 8
  for (int i = 0; i < 4; ++i)
    tile[ty + i * 8][tx] = in[(size_t)(br + ty + i * 8) * cols + bc + tx];
  __syncthreads();
  for (int i = 0; i < 4; ++i)
    out[(size_t)(bc + ty + i * 8) * rows + br + tx] = f2bf(tile[tx][ty + i * 8]);
}

// ---------------------------------------------------------------------------
// RMSNorm -> bf16 output
// ---------------------------------------------------------------------------
__global__ void __launch_bounds__(256)
rmsnorm_kernel(const float* __restrict__ x, const float* __restrict__ g,
               uint16_t* __restrict__ y) {
  __shared__ float red[8];
  const int row = blockIdx.x;
  const float* xr = x + (size_t)row * DIM;
  float ss = 0.f;
  for (int i = threadIdx.x; i < DIM; i += 256) { float v = xr[i]; ss += v * v; }
  for (int m = 1; m < 32; m <<= 1) ss += __shfl_xor(ss, m, 32);
  if ((threadIdx.x & 31) == 0) red[threadIdx.x >> 5] = ss;
  __syncthreads();
  if (threadIdx.x < 8) {
    float v = red[threadIdx.x];
    for (int m = 1; m < 8; m <<= 1) v += __shfl_xor(v, m, 32);
    if (threadIdx.x == 0) red[0] = v;
  }
  __syncthreads();
  const float denom = rsqrtf(red[0] / (float)DIM + 1e-6f);
  for (int i = threadIdx.x; i < DIM; i += 256)
    y[(size_t)row * DIM + i] = f2bf(xr[i] * denom * g[i]);
}

// ---------------------------------------------------------------------------
// Rotary (interleaved pairs). fp32 in-place (Q) / fp32 -> bf16 (K).
// ---------------------------------------------------------------------------
__global__ void rotary_q_kernel(float* __restrict__ x,
                                const float* __restrict__ cosb,
                                const float* __restrict__ sinb, int total) {
  int idx = blockIdx.x * blockDim.x + threadIdx.x;
  if (idx >= total) return;
  int d2 = idx & 63, rem = idx >> 6;
  int h = rem % N_HEADS, s = rem / N_HEADS;
  float c = cosb[s * 64 + d2], sn = sinb[s * 64 + d2];
  float* p = x + ((size_t)s * N_HEADS + h) * HEAD_DIM + 2 * d2;
  float re = p[0], im = p[1];
  p[0] = re * c - im * sn;
  p[1] = re * sn + im * c;
}
__global__ void rotary_k_kernel(const float* __restrict__ kin,
                                uint16_t* __restrict__ kout,
                                const float* __restrict__ cosb,
                                const float* __restrict__ sinb, int total) {
  int idx = blockIdx.x * blockDim.x + threadIdx.x;
  if (idx >= total) return;
  int d2 = idx & 63, rem = idx >> 6;
  int h = rem % N_KV_HEADS, s = rem / N_KV_HEADS;
  float c = cosb[s * 64 + d2], sn = sinb[s * 64 + d2];
  size_t base = ((size_t)s * N_KV_HEADS + h) * HEAD_DIM + 2 * d2;
  float re = kin[base], im = kin[base + 1];
  kout[base]     = f2bf(re * c - im * sn);
  kout[base + 1] = f2bf(re * sn + im * c);
}

// ---------------------------------------------------------------------------
// bf16 GEMM, async double-buffered LDS: C[M,N] = A[M,K] * Bt[N,K]^T
// Block tile 128x64x32; 4 waves; wave = 32x64 (2x4 WMMA frags).
// Epilogue: optional fp32 out (+residual) and/or bf16 out.
// ---------------------------------------------------------------------------
#define TSTRIDE 40   // padded LDS row stride (bf16 elems); 16B-aligned chunks
__global__ void __launch_bounds__(128)
gemm_bf16_kernel(const uint16_t* __restrict__ A, const uint16_t* __restrict__ Bt,
                 const float* __restrict__ R, float* __restrict__ Cf,
                 uint16_t* __restrict__ Cb, int M, int N, int K) {
  __shared__ __attribute__((aligned(16))) uint16_t As[2][128 * TSTRIDE];
  __shared__ __attribute__((aligned(16))) uint16_t Bs[2][64 * TSTRIDE];
  const int t = threadIdx.x;
  const int wave = t >> 5, lane = t & 31;
  const int hm = lane & 15, half = lane >> 4;
  const int n0 = blockIdx.x * 64, m0 = blockIdx.y * 128;
  const int wm = wave * 32;

  auto issue = [&](int buf, int kb) {
    for (int i = 0; i < 4; ++i) {               // A: 128x32 = 512 chunks
      int c = t + i * 128;
      int m = c >> 2, cc = (c & 3) * 8;
      async_b128(lds_off32(&As[buf][m * TSTRIDE + cc]),
                 A + (size_t)(m0 + m) * K + kb + cc);
    }
    for (int i = 0; i < 2; ++i) {               // B: 64x32 = 256 chunks
      int c = t + i * 128;
      int n = c >> 2, cc = (c & 3) * 8;
      async_b128(lds_off32(&Bs[buf][n * TSTRIDE + cc]),
                 Bt + (size_t)(n0 + n) * K + kb + cc);
    }
  };

  v8f acc[2][4] = {};
  issue(0, 0);
  int buf = 0;
  for (int kb = 0; kb < K; kb += 32) {
    if (kb + 32 < K) { issue(buf ^ 1, kb + 32); WAIT_ASYNC(0x6); }
    else             { WAIT_ASYNC(0x0); }
    __syncthreads();
    Frag16 af[2], bf[4];
    for (int mf = 0; mf < 2; ++mf) {
      int m = wm + mf * 16 + hm;
      for (int v = 0; v < 8; ++v)
        af[mf].u[v] = *(const uint32_t*)&As[buf][m * TSTRIDE + a_k(v, half)];
    }
    for (int nf = 0; nf < 4; ++nf) {
      int n = nf * 16 + hm;
      for (int v = 0; v < 8; ++v)
        bf[nf].u[v] = *(const uint32_t*)&Bs[buf][n * TSTRIDE + half * 16 + 2 * v];
    }
    for (int mf = 0; mf < 2; ++mf)
      for (int nf = 0; nf < 4; ++nf)
        acc[mf][nf] = __builtin_amdgcn_wmma_f32_16x16x32_bf16(
            false, af[mf].v, false, bf[nf].v, (short)0, acc[mf][nf], false, false);
    __syncthreads();
    buf ^= 1;
  }
  for (int mf = 0; mf < 2; ++mf)
    for (int nf = 0; nf < 4; ++nf)
      for (int r = 0; r < 8; ++r) {
        int row = m0 + wm + mf * 16 + r + 8 * half;
        int col = n0 + nf * 16 + hm;
        float v = acc[mf][nf][r];
        if (R)  v += R[(size_t)row * N + col];
        if (Cf) Cf[(size_t)row * N + col] = v;
        if (Cb) Cb[(size_t)row * N + col] = f2bf(v);
      }
}

// ---------------------------------------------------------------------------
// Fused FFN gate GEMM: Gb = bf16( silu(A*B1t^T) * (A*B3t^T) )
// ---------------------------------------------------------------------------
__global__ void __launch_bounds__(128)
gemm_gate_kernel(const uint16_t* __restrict__ A, const uint16_t* __restrict__ B1t,
                 const uint16_t* __restrict__ B3t, uint16_t* __restrict__ Gb,
                 int M, int N, int K) {
  __shared__ __attribute__((aligned(16))) uint16_t As[2][128 * TSTRIDE];
  __shared__ __attribute__((aligned(16))) uint16_t B1s[2][64 * TSTRIDE];
  __shared__ __attribute__((aligned(16))) uint16_t B3s[2][64 * TSTRIDE];
  const int t = threadIdx.x;
  const int wave = t >> 5, lane = t & 31;
  const int hm = lane & 15, half = lane >> 4;
  const int n0 = blockIdx.x * 64, m0 = blockIdx.y * 128;
  const int wm = wave * 32;

  auto issue = [&](int buf, int kb) {
    for (int i = 0; i < 4; ++i) {
      int c = t + i * 128;
      int m = c >> 2, cc = (c & 3) * 8;
      async_b128(lds_off32(&As[buf][m * TSTRIDE + cc]),
                 A + (size_t)(m0 + m) * K + kb + cc);
    }
    for (int i = 0; i < 2; ++i) {
      int c = t + i * 128;
      int n = c >> 2, cc = (c & 3) * 8;
      async_b128(lds_off32(&B1s[buf][n * TSTRIDE + cc]),
                 B1t + (size_t)(n0 + n) * K + kb + cc);
      async_b128(lds_off32(&B3s[buf][n * TSTRIDE + cc]),
                 B3t + (size_t)(n0 + n) * K + kb + cc);
    }
  };

  v8f acc1[2][4] = {}, acc3[2][4] = {};
  issue(0, 0);
  int buf = 0;
  for (int kb = 0; kb < K; kb += 32) {
    if (kb + 32 < K) { issue(buf ^ 1, kb + 32); WAIT_ASYNC(0x8); }
    else             { WAIT_ASYNC(0x0); }
    __syncthreads();
    Frag16 af[2], b1[4], b3[4];
    for (int mf = 0; mf < 2; ++mf) {
      int m = wm + mf * 16 + hm;
      for (int v = 0; v < 8; ++v)
        af[mf].u[v] = *(const uint32_t*)&As[buf][m * TSTRIDE + a_k(v, half)];
    }
    for (int nf = 0; nf < 4; ++nf) {
      int n = nf * 16 + hm;
      for (int v = 0; v < 8; ++v) {
        b1[nf].u[v] = *(const uint32_t*)&B1s[buf][n * TSTRIDE + half * 16 + 2 * v];
        b3[nf].u[v] = *(const uint32_t*)&B3s[buf][n * TSTRIDE + half * 16 + 2 * v];
      }
    }
    for (int mf = 0; mf < 2; ++mf)
      for (int nf = 0; nf < 4; ++nf) {
        acc1[mf][nf] = __builtin_amdgcn_wmma_f32_16x16x32_bf16(
            false, af[mf].v, false, b1[nf].v, (short)0, acc1[mf][nf], false, false);
        acc3[mf][nf] = __builtin_amdgcn_wmma_f32_16x16x32_bf16(
            false, af[mf].v, false, b3[nf].v, (short)0, acc3[mf][nf], false, false);
      }
    __syncthreads();
    buf ^= 1;
  }
  for (int mf = 0; mf < 2; ++mf)
    for (int nf = 0; nf < 4; ++nf)
      for (int r = 0; r < 8; ++r) {
        int row = m0 + wm + mf * 16 + r + 8 * half;
        int col = n0 + nf * 16 + hm;
        float a = acc1[mf][nf][r];
        float s = a / (1.f + __expf(-a));
        Gb[(size_t)row * N + col] = f2bf(s * acc3[mf][nf][r]);
      }
}

// ---------------------------------------------------------------------------
// Flash attention (causal, GQA n_rep=4). Block: 4 waves, 64 q-rows, 1 head.
// K panel from pre-rotated bf16 K; V panel from pre-transposed bf16 V[d][s].
// Panels DMA'd with async-to-LDS. Output bf16.
// ---------------------------------------------------------------------------
#define KSTRIDE 136
#define VSTRIDE 40
__global__ void __launch_bounds__(128)
attn_kernel(const float* __restrict__ Q, const uint16_t* __restrict__ Kb,
            const uint16_t* __restrict__ Vt, uint16_t* __restrict__ Ob) {
  __shared__ __attribute__((aligned(16))) uint16_t Ks[32 * KSTRIDE];  // [kpos][d]
  __shared__ __attribute__((aligned(16))) uint16_t Vs[128 * VSTRIDE]; // [d][kpos]
  __shared__ __attribute__((aligned(16))) uint16_t Ps[4][16][36];
  const int t = threadIdx.x, wave = t >> 5, lane = t & 31;
  const int hm = lane & 15, half = lane >> 4;
  const int h = blockIdx.y;
  const int qb0 = blockIdx.x * 64;
  const int kvh = h >> 2;
  const int qrow0 = qb0 + wave * 16;
  const float scale = 0.08838834764831845f;   // 1/sqrt(128)

  Frag16 qf[4];
  {
    const float* qp = Q + (size_t)(qrow0 + hm) * DIM + h * HEAD_DIM;
    for (int dc = 0; dc < 4; ++dc)
      for (int v = 0; v < 8; ++v) {
        int d = dc * 32 + a_k(v, half);
        qf[dc].u[v] = pack2bf(qp[d], qp[d + 1]);
      }
  }
  v8f o[8] = {};
  float mrow[8], lrow[8];
  for (int r = 0; r < 8; ++r) { mrow[r] = -1e30f; lrow[r] = 0.f; }

  const int kend = qb0 + 64;
  for (int kb = 0; kb < kend; kb += 32) {
    for (int i = 0; i < 4; ++i) {               // K panel: 32x128 = 512 chunks
      int c = t + i * 128;
      int kk = c >> 4, cc = (c & 15) * 8;
      async_b128(lds_off32(&Ks[kk * KSTRIDE + cc]),
                 Kb + (size_t)(kb + kk) * KV_DIM + kvh * HEAD_DIM + cc);
    }
    for (int i = 0; i < 4; ++i) {               // V panel: 128x32 = 512 chunks
      int c = t + i * 128;
      int d = c >> 2, cc = (c & 3) * 8;
      async_b128(lds_off32(&Vs[d * VSTRIDE + cc]),
                 Vt + (size_t)(kvh * HEAD_DIM + d) * SEQ + kb + cc);
    }
    WAIT_ASYNC(0x0);
    __syncthreads();

    v8f s[2] = {};
    for (int sf = 0; sf < 2; ++sf) {
      int n = sf * 16 + hm;
      for (int dc = 0; dc < 4; ++dc) {
        Frag16 kf;
        for (int v = 0; v < 8; ++v)
          kf.u[v] = *(const uint32_t*)&Ks[n * KSTRIDE + dc * 32 + half * 16 + 2 * v];
        s[sf] = __builtin_amdgcn_wmma_f32_16x16x32_bf16(
            false, qf[dc].v, false, kf.v, (short)0, s[sf], false, false);
      }
    }
    float bmax[8];
    for (int r = 0; r < 8; ++r) bmax[r] = -1e30f;
    for (int sf = 0; sf < 2; ++sf)
      for (int r = 0; r < 8; ++r) {
        int qi = qrow0 + r + 8 * half;
        int kj = kb + sf * 16 + hm;
        float v = s[sf][r] * scale;
        if (kj > qi) v = -1e30f;
        s[sf][r] = v;
        bmax[r] = fmaxf(bmax[r], v);
      }
    for (int m = 1; m < 16; m <<= 1)
      for (int r = 0; r < 8; ++r)
        bmax[r] = fmaxf(bmax[r], __shfl_xor(bmax[r], m, 32));
    float corr[8], rs[8];
    for (int r = 0; r < 8; ++r) {
      float nm = fmaxf(mrow[r], bmax[r]);
      corr[r] = __expf(mrow[r] - nm);
      mrow[r] = nm;
      rs[r] = 0.f;
    }
    for (int sf = 0; sf < 2; ++sf)
      for (int r = 0; r < 8; ++r) {
        float p = __expf(s[sf][r] - mrow[r]);
        s[sf][r] = p;
        rs[r] += p;
      }
    for (int m = 1; m < 16; m <<= 1)
      for (int r = 0; r < 8; ++r)
        rs[r] += __shfl_xor(rs[r], m, 32);
    for (int r = 0; r < 8; ++r) lrow[r] = lrow[r] * corr[r] + rs[r];
    for (int nf = 0; nf < 8; ++nf)
      for (int r = 0; r < 8; ++r) o[nf][r] *= corr[r];

    for (int sf = 0; sf < 2; ++sf)
      for (int r = 0; r < 8; ++r)
        Ps[wave][r + 8 * half][sf * 16 + hm] = f2bf(s[sf][r]);
    __syncthreads();
    Frag16 pf;
    for (int v = 0; v < 8; ++v)
      pf.u[v] = *(const uint32_t*)&Ps[wave][hm][a_k(v, half)];

    for (int nf = 0; nf < 8; ++nf) {
      Frag16 vf;
      int d = nf * 16 + hm;
      for (int v = 0; v < 8; ++v)
        vf.u[v] = *(const uint32_t*)&Vs[d * VSTRIDE + half * 16 + 2 * v];
      o[nf] = __builtin_amdgcn_wmma_f32_16x16x32_bf16(
          false, pf.v, false, vf.v, (short)0, o[nf], false, false);
    }
    __syncthreads();
  }
  for (int nf = 0; nf < 8; ++nf)
    for (int r = 0; r < 8; ++r) {
      int row = qrow0 + r + 8 * half;
      int col = h * HEAD_DIM + nf * 16 + hm;
      Ob[(size_t)row * DIM + col] = f2bf(o[nf][r] / lrow[r]);
    }
}

// ---------------------------------------------------------------------------
extern "C" void kernel_launch(void* const* d_in, const int* in_sizes, int n_in,
                              void* d_out, int out_size, void* d_ws, size_t ws_size,
                              hipStream_t stream) {
  const float* x  = (const float*)d_in[0];
  const float* fc = (const float*)d_in[1];
  const float* fs = (const float*)d_in[2];
  const float* wq = (const float*)d_in[4];
  const float* wk = (const float*)d_in[5];
  const float* wv = (const float*)d_in[6];
  const float* wo = (const float*)d_in[7];
  const float* w1 = (const float*)d_in[8];
  const float* w2 = (const float*)d_in[9];
  const float* w3 = (const float*)d_in[10];
  const float* ga = (const float*)d_in[11];
  const float* gf = (const float*)d_in[12];
  float* out = (float*)d_out;

  char* cur = (char*)d_ws;
  auto alloc = [&](size_t bytes) -> void* {
    void* r = (void*)cur;
    cur += (bytes + 255) & ~(size_t)255;
    return r;
  };
  uint16_t* wqT = (uint16_t*)alloc((size_t)DIM * DIM * 2);
  uint16_t* wkT = (uint16_t*)alloc((size_t)DIM * KV_DIM * 2);
  uint16_t* wvT = (uint16_t*)alloc((size_t)DIM * KV_DIM * 2);
  uint16_t* woT = (uint16_t*)alloc((size_t)DIM * DIM * 2);
  uint16_t* w1T = (uint16_t*)alloc((size_t)DIM * HIDDEN * 2);
  uint16_t* w2T = (uint16_t*)alloc((size_t)DIM * HIDDEN * 2);
  uint16_t* w3T = (uint16_t*)alloc((size_t)DIM * HIDDEN * 2);
  uint16_t* xnb = (uint16_t*)alloc((size_t)SEQ * DIM * 2);     // rmsnorm out (reused)
  float*    q   = (float*)   alloc((size_t)SEQ * DIM * 4);
  float*    k   = (float*)   alloc((size_t)SEQ * KV_DIM * 4);
  float*    v   = (float*)   alloc((size_t)SEQ * KV_DIM * 4);
  uint16_t* kb  = (uint16_t*)alloc((size_t)SEQ * KV_DIM * 2);  // rotated K, bf16
  uint16_t* vT  = (uint16_t*)alloc((size_t)SEQ * KV_DIM * 2);  // V^T [d][s], bf16
  uint16_t* atb = (uint16_t*)alloc((size_t)SEQ * DIM * 2);     // attention out, bf16
  float*    hb  = (float*)   alloc((size_t)SEQ * DIM * 4);     // post-attn residual
  uint16_t* g1b = (uint16_t*)alloc((size_t)SEQ * HIDDEN * 2);  // gated FFN, bf16

  const dim3 blk128(128), blk256(256);

  // 0) weight convert+transpose (bf16, K-contiguous; FFN weights fit L2)
  transpose_bf16_kernel<<<dim3(DIM / 32, DIM / 32), blk256, 0, stream>>>(wq, wqT, DIM, DIM);
  transpose_bf16_kernel<<<dim3(KV_DIM / 32, DIM / 32), blk256, 0, stream>>>(wk, wkT, DIM, KV_DIM);
  transpose_bf16_kernel<<<dim3(KV_DIM / 32, DIM / 32), blk256, 0, stream>>>(wv, wvT, DIM, KV_DIM);
  transpose_bf16_kernel<<<dim3(DIM / 32, DIM / 32), blk256, 0, stream>>>(wo, woT, DIM, DIM);
  transpose_bf16_kernel<<<dim3(HIDDEN / 32, DIM / 32), blk256, 0, stream>>>(w1, w1T, DIM, HIDDEN);
  transpose_bf16_kernel<<<dim3(DIM / 32, HIDDEN / 32), blk256, 0, stream>>>(w2, w2T, HIDDEN, DIM);
  transpose_bf16_kernel<<<dim3(HIDDEN / 32, DIM / 32), blk256, 0, stream>>>(w3, w3T, DIM, HIDDEN);

  // 1) attention rmsnorm -> bf16
  rmsnorm_kernel<<<SEQ, blk256, 0, stream>>>(x, ga, xnb);
  // 2) Q/K/V projections (fp32 out)
  gemm_bf16_kernel<<<dim3(DIM / 64, SEQ / 128), blk128, 0, stream>>>(
      xnb, wqT, nullptr, q, nullptr, SEQ, DIM, DIM);
  gemm_bf16_kernel<<<dim3(KV_DIM / 64, SEQ / 128), blk128, 0, stream>>>(
      xnb, wkT, nullptr, k, nullptr, SEQ, KV_DIM, DIM);
  gemm_bf16_kernel<<<dim3(KV_DIM / 64, SEQ / 128), blk128, 0, stream>>>(
      xnb, wvT, nullptr, v, nullptr, SEQ, KV_DIM, DIM);
  // 3) rotary: Q in-place fp32; K -> bf16. V -> bf16 transposed [d][s].
  {
    int tq = SEQ * N_HEADS * (HEAD_DIM / 2);
    int tk = SEQ * N_KV_HEADS * (HEAD_DIM / 2);
    rotary_q_kernel<<<(tq + 255) / 256, blk256, 0, stream>>>(q, fc, fs, tq);
    rotary_k_kernel<<<(tk + 255) / 256, blk256, 0, stream>>>(k, kb, fc, fs, tk);
    transpose_bf16_kernel<<<dim3(KV_DIM / 32, SEQ / 32), blk256, 0, stream>>>(v, vT, SEQ, KV_DIM);
  }
  // 4) flash attention -> bf16
  attn_kernel<<<dim3(SEQ / 64, N_HEADS), blk128, 0, stream>>>(q, kb, vT, atb);
  // 5) output projection + residual: hb = x + atb @ wo
  gemm_bf16_kernel<<<dim3(DIM / 64, SEQ / 128), blk128, 0, stream>>>(
      atb, woT, x, hb, nullptr, SEQ, DIM, DIM);
  // 6) ffn rmsnorm -> bf16 (reuse xnb)
  rmsnorm_kernel<<<SEQ, blk256, 0, stream>>>(hb, gf, xnb);
  // 7) fused gate: g1b = bf16(silu(xnb@w1) * (xnb@w3))
  gemm_gate_kernel<<<dim3(HIDDEN / 64, SEQ / 128), blk128, 0, stream>>>(
      xnb, w1T, w3T, g1b, SEQ, HIDDEN, DIM);
  // 8) down projection + residual: out = hb + g1b @ w2
  gemm_bf16_kernel<<<dim3(DIM / 64, SEQ / 128), blk128, 0, stream>>>(
      g1b, w2T, hb, out, nullptr, SEQ, DIM, HIDDEN);
}
